// TemplatePairStackBlock_20993800143249
// MI455X (gfx1250) — matrix operand order repaired and checked by reading, not compiled, not run on hardware
//
#include <hip/hip_runtime.h>
#include <hip/hip_bf16.h>

// ============================================================================
// TemplatePairStackBlock on MI455X (gfx1250, wave32, WMMA).
//
// ~36 GFLOP; HBM-light (~300MB) via fused flash attention (scores never hit
// HBM). All matmul einsums on v_wmma_f32_16x16x32_f16 (f16 in, f32 acc).
// LDS staging uses gfx1250 async loads (GLOBAL_LOAD_ASYNC_TO_LDS_B128,
// ASYNCcnt) — builtin present; signature (global v4i*, lds v4i*, imm, imm).
// GEMM is templated on (K, NOUT, FLAGS) so inner loops and epilogues are
// branch-free at runtime.
//
// Input flattening assumption: jax.tree_util.tree_leaves order
// (dict keys sorted at every nesting level). See enum below.
// Workspace: needs ~94 MB. d_out doubles as the f32 residual buffer.
// ============================================================================

typedef __attribute__((ext_vector_type(16))) _Float16 v16h;
typedef __attribute__((ext_vector_type(8)))  _Float16 v8h;
typedef __attribute__((ext_vector_type(8)))  float    v8f;
typedef __attribute__((ext_vector_type(4)))  int      v4i;
typedef _Float16 half_t;

#define NRES 256
#define CZ   64
#define CH   32
#define NH   4
#define M_TOT (NRES*NRES)   // 65536

#ifndef __has_builtin
#define __has_builtin(x) 0
#endif

// --------------------------- async LDS staging ------------------------------
#if __has_builtin(__builtin_amdgcn_global_load_async_to_lds_b128)
#define HAVE_ASYNC_LDS 1
#else
#define HAVE_ASYNC_LDS 0
#endif

__device__ __forceinline__ void cp_async16(half_t* lds_dst, const half_t* gsrc) {
#if HAVE_ASYNC_LDS
  __builtin_amdgcn_global_load_async_to_lds_b128(
      (__attribute__((address_space(1))) v4i*)gsrc,
      (__attribute__((address_space(3))) v4i*)lds_dst,
      /*offset=*/0, /*cpol=*/0);
#else
  *(v8h*)lds_dst = *(const v8h*)gsrc;
#endif
}

__device__ __forceinline__ void async_wait0() {
#if HAVE_ASYNC_LDS
  asm volatile("s_wait_asynccnt 0" ::: "memory");
#endif
}

// ------------------------------- WMMA helpers -------------------------------

__device__ __forceinline__ v8f wmma16(v16h a, v16h b, v8f c) {
  // D(16x16,f32) = A(16x32,f16) x B(32x16,f16) + C
  return __builtin_amdgcn_wmma_f32_16x16x32_f16(
      /*neg_a=*/false, a, /*neg_b=*/false, b,
      /*c_mod=*/(short)0, c, /*reuse_a=*/false, /*reuse_b=*/false);
}

// A fragment (16x32 f16) from row-major [16 x 32] tile, row stride ld halfs.
// Lane l holds row (l&15); element e holds K = (e<8 ? e : 8+e) + 8*(l>>4)
// -> two contiguous 16-byte runs (b128 loads).
__device__ __forceinline__ v16h load_a_frag(const half_t* __restrict__ p, int ld) {
  int lane = threadIdx.x & 31;
  const half_t* r = p + (size_t)(lane & 15) * ld + (lane >> 4) * 8;
  v8h lo = *(const v8h*)r;
  v8h hi = *(const v8h*)(r + 16);
  v16h a;
#pragma unroll
  for (int t = 0; t < 8; ++t) { a[t] = lo[t]; a[8 + t] = hi[t]; }
  return a;
}

// B fragment (32x16 f16) from N-major storage Bt[n][k], row stride ld halfs.
// Lane l holds col n=(l&15); element e holds K = (e<8 ? e : 8+e) + 8*(l>>4)
// -> two contiguous 16-byte runs (b128 loads).
__device__ __forceinline__ v16h load_b_frag_nk(const half_t* __restrict__ p, int ld) {
  int lane = threadIdx.x & 31;
  const half_t* r = p + (size_t)(lane & 15) * ld + (lane >> 4) * 8;
  v8h lo = *(const v8h*)r;
  v8h hi = *(const v8h*)(r + 16);
  v16h b;
#pragma unroll
  for (int t = 0; t < 8; ++t) { b[t] = lo[t]; b[8 + t] = hi[t]; }
  return b;
}

__device__ __forceinline__ float sigmoidf_(float x) {
  return 1.f / (1.f + __expf(-x));
}

// ------------------------------ weight packing ------------------------------

// Wq/Wk/Wv/Wg [H,CZ,C] f32 -> f16 Bt layout [512 x 64]:
// row = which*128 + h*32 + c, col = d, value = W[h][d][c]
__global__ void pack_qkvg(const float* __restrict__ wq, const float* __restrict__ wk,
                          const float* __restrict__ wv, const float* __restrict__ wg,
                          half_t* __restrict__ out) {
  int t = blockIdx.x * 256 + threadIdx.x;
  if (t >= 512 * 64) return;
  int row = t >> 6, d = t & 63;
  int which = row >> 7, hc = row & 127, h = hc >> 5, c = hc & 31;
  const float* W = which == 0 ? wq : which == 1 ? wk : which == 2 ? wv : wg;
  out[t] = (half_t)W[h * 2048 + d * 32 + c];
}

// Wo [H,C,CZ] -> Bt [64 x 128]: row = dd, col = h*32+c, value = wo[h][c][dd]
__global__ void pack_wo_attn(const float* __restrict__ wo, half_t* __restrict__ out) {
  int t = blockIdx.x * 256 + threadIdx.x;
  if (t >= 64 * 128) return;
  int row = t >> 7, hc = t & 127, h = hc >> 5, c = hc & 31;
  out[t] = (half_t)wo[h * 2048 + c * 64 + row];
}

// wag/wap/wbg/wbp [CZ,C] -> Bt [128 x 64] (rows: ag|ap|bg|bp), plus bias128
__global__ void pack_ab(const float* __restrict__ wag, const float* __restrict__ wap,
                        const float* __restrict__ wbg, const float* __restrict__ wbp,
                        const float* __restrict__ bag, const float* __restrict__ bap,
                        const float* __restrict__ bbg, const float* __restrict__ bbp,
                        half_t* __restrict__ out, float* __restrict__ bias128) {
  int t = blockIdx.x * 256 + threadIdx.x;
  if (t >= 128 * 64) return;
  int row = t >> 6, d = t & 63, which = row >> 5, c = row & 31;
  const float* W = which == 0 ? wag : which == 1 ? wap : which == 2 ? wbg : wbp;
  out[t] = (half_t)W[d * 32 + c];
  if (d == 0) {
    const float* B = which == 0 ? bag : which == 1 ? bap : which == 2 ? bbg : bbp;
    bias128[row] = B[c];
  }
}

// generic [Kdim x Rdim] f32 -> f16 Bt [Rdim x Kdim]: out[r*K+k] = in[k*R+r]
__global__ void transpose_pack(const float* __restrict__ in, half_t* __restrict__ out,
                               int Kdim, int Rdim) {
  int t = blockIdx.x * 256 + threadIdx.x;
  if (t >= Kdim * Rdim) return;
  int r = t / Kdim, k = t % Kdim;
  out[t] = (half_t)in[k * Rdim + r];
}

// ------------------------------- layer norms --------------------------------

__global__ __launch_bounds__(256)
void ln64_f16(const float* __restrict__ x, const float* __restrict__ g,
              const float* __restrict__ b, half_t* __restrict__ out, int transpose) {
  int row = blockIdx.x * 8 + (threadIdx.x >> 5);
  int lane = threadIdx.x & 31;
  int src = transpose ? (((row & 255) << 8) | (row >> 8)) : row;
  float v0 = x[(size_t)src * 64 + lane];
  float v1 = x[(size_t)src * 64 + 32 + lane];
  float s = v0 + v1;
#pragma unroll
  for (int d = 1; d < 32; d <<= 1) s += __shfl_xor(s, d, 32);
  float mean = s * (1.f / 64.f);
  float d0 = v0 - mean, d1 = v1 - mean;
  float q = d0 * d0 + d1 * d1;
#pragma unroll
  for (int d = 1; d < 32; d <<= 1) q += __shfl_xor(q, d, 32);
  float inv = rsqrtf(q * (1.f / 64.f) + 1e-5f);
  out[(size_t)row * 64 + lane]      = (half_t)(d0 * inv * g[lane] + b[lane]);
  out[(size_t)row * 64 + 32 + lane] = (half_t)(d1 * inv * g[32 + lane] + b[32 + lane]);
}

__global__ __launch_bounds__(256)
void ln32_f16(const float* __restrict__ z, const float* __restrict__ g,
              const float* __restrict__ b, half_t* __restrict__ out) {
  int row = blockIdx.x * 8 + (threadIdx.x >> 5);
  int lane = threadIdx.x & 31;
  float v = z[(size_t)row * 32 + lane];
  float s = v;
#pragma unroll
  for (int d = 1; d < 32; d <<= 1) s += __shfl_xor(s, d, 32);
  float mean = s * (1.f / 32.f);
  float dd = v - mean;
  float q = dd * dd;
#pragma unroll
  for (int d = 1; d < 32; d <<= 1) q += __shfl_xor(q, d, 32);
  float inv = rsqrtf(q * (1.f / 32.f) + 1e-5f);
  out[(size_t)row * 32 + lane] = (half_t)(dd * inv * g[lane] + b[lane]);
}

__global__ __launch_bounds__(256)
void final_ln64(float* __restrict__ x, const float* __restrict__ g,
                const float* __restrict__ b) {
  int row = blockIdx.x * 8 + (threadIdx.x >> 5);
  int lane = threadIdx.x & 31;
  float v0 = x[(size_t)row * 64 + lane];
  float v1 = x[(size_t)row * 64 + 32 + lane];
  float s = v0 + v1;
#pragma unroll
  for (int d = 1; d < 32; d <<= 1) s += __shfl_xor(s, d, 32);
  float mean = s * (1.f / 64.f);
  float d0 = v0 - mean, d1 = v1 - mean;
  float q = d0 * d0 + d1 * d1;
#pragma unroll
  for (int d = 1; d < 32; d <<= 1) q += __shfl_xor(q, d, 32);
  float inv = rsqrtf(q * (1.f / 64.f) + 1e-5f);
  x[(size_t)row * 64 + lane]      = d0 * inv * g[lane] + b[lane];
  x[(size_t)row * 64 + 32 + lane] = d1 * inv * g[32 + lane] + b[32 + lane];
}

// ------------------------ generic WMMA GEMM + epilogue ----------------------
// out[m, col0+n] = epilogue( sum_k A[m,k] * Bt[n,k] + bias[n] ), lda == K.
// Block owns a 128-row A slab: staged once into LDS (async), A fragments kept
// in registers, then all N tiles swept with B streamed through L2.
// K/NOUT/FLAGS are template params -> branch-free inner loop and epilogue.
enum GemmFlags {
  F_BIAS = 1, F_SIG = 2, F_RELU = 4,
  F_OUT16 = 8,        // store f16 to out16
  F_OUT32 = 16,       // store f32 to out32
  F_RES = 32,         // out32[...] += v  (residual accumulate)
  F_GATE = 64,        // v *= gate[m*ld_out+col0+n]
  F_REST = 128        // residual index transposed: m=(i,j) -> (j,i)
};

template <int K, int NOUT, int FLAGS>
__global__ __launch_bounds__(256)
void gemm_nk(const half_t* __restrict__ A,     // [M x K], contiguous (lda==K)
             const half_t* __restrict__ Bt,    // [NOUT x K]
             const float* __restrict__ bias,
             half_t* __restrict__ out16, float* __restrict__ out32,
             const float* __restrict__ gate,
             int ld_out, int col0) {
  constexpr int KSTEPS = K / 32;
  constexpr int NTILES = NOUT / 16;
  __shared__ __align__(16) half_t As[128 * K];
  int tid = threadIdx.x, lane = tid & 31, w = tid >> 5;
  int m0 = blockIdx.x * 128;

  // async-stage the contiguous A slab [128 x K]
  const half_t* Ablk = A + (size_t)m0 * K;
  constexpr int CHUNKS = (128 * K) / 8;   // 16-byte chunks
#pragma unroll
  for (int t = 0; t < CHUNKS / 256; ++t)
    cp_async16(&As[(t * 256 + tid) * 8], Ablk + (size_t)(t * 256 + tid) * 8);
  async_wait0();
  __syncthreads();

  // per-wave A fragments for rows [w*16, w*16+16), all K steps in registers
  v16h af[KSTEPS];
#pragma unroll
  for (int ks = 0; ks < KSTEPS; ++ks)
    af[ks] = load_a_frag(As + (size_t)(w * 16) * K + ks * 32, K);

  int half_ = lane >> 4, ln = lane & 15;
  for (int nt = 0; nt < NTILES; ++nt) {
    int n0 = nt * 16;
    if (nt + 1 < NTILES)  // gfx1250 global_prefetch_b8 of next B tile
      __builtin_prefetch(Bt + (size_t)(n0 + 16) * K, 0, 3);
    v8f acc = {0.f, 0.f, 0.f, 0.f, 0.f, 0.f, 0.f, 0.f};
#pragma unroll
    for (int ks = 0; ks < KSTEPS; ++ks)
      acc = wmma16(af[ks], load_b_frag_nk(Bt + (size_t)n0 * K + ks * 32, K), acc);

    int n = n0 + ln;
    float bv = 0.f;
    if constexpr (FLAGS & F_BIAS) bv = bias[n];
#pragma unroll
    for (int r = 0; r < 8; ++r) {
      int m = m0 + w * 16 + r + 8 * half_;
      float v = acc[r] + bv;
      if constexpr (FLAGS & F_SIG)  v = sigmoidf_(v);
      if constexpr (FLAGS & F_RELU) v = fmaxf(v, 0.f);
      if constexpr (FLAGS & F_GATE) v *= gate[(size_t)m * ld_out + col0 + n];
      size_t oi = (size_t)m * ld_out + col0 + n;
      if constexpr (FLAGS & F_OUT16) out16[oi] = (half_t)v;
      if constexpr (FLAGS & F_OUT32) out32[oi] = v;
      if constexpr (FLAGS & F_RES) {
        size_t mm = (FLAGS & F_REST) ? (size_t)((m & 255) * 256 + (m >> 8)) : (size_t)m;
        out32[mm * ld_out + col0 + n] += v;
      }
    }
  }
}

// --------------------------- attention pair bias ----------------------------
// biasw[h][i][j] = sum_d xl[(i,j),d] * wb[h][d]
__global__ __launch_bounds__(256)
void bias_proj(const half_t* __restrict__ xl, const float* __restrict__ wb,
               float* __restrict__ biasw) {
  int m = blockIdx.x * 256 + threadIdx.x;
  float acc0 = 0.f, acc1 = 0.f, acc2 = 0.f, acc3 = 0.f;
  const half_t* xr = xl + (size_t)m * 64;
#pragma unroll 8
  for (int d = 0; d < 64; ++d) {
    float xv = (float)xr[d];
    acc0 += xv * wb[d];
    acc1 += xv * wb[64 + d];
    acc2 += xv * wb[128 + d];
    acc3 += xv * wb[192 + d];
  }
  biasw[0 * M_TOT + m] = acc0;
  biasw[1 * M_TOT + m] = acc1;
  biasw[2 * M_TOT + m] = acc2;
  biasw[3 * M_TOT + m] = acc3;
}

// ------------------------------ flash attention -----------------------------
// One block per (h, i): softmax_k( q_ij.k_ik/sqrt(C) + bias_jk ) @ v, gated.
// proj row m=(i,j): [q(0..127) | k(128..255) | v(256..383) | g(384..511)],
// inner index h*32+c; g already sigmoided. Output go[m, h*32+c] f16.
__global__ __launch_bounds__(256)
void tri_attn_fa(const half_t* __restrict__ proj, const float* __restrict__ biasw,
                 half_t* __restrict__ go) {
  __shared__ __align__(16) half_t Ks[NRES * 32];   // [k_idx][c]  (= Bt for QK^T)
  __shared__ __align__(16) half_t Vt[32 * NRES];   // [c][k_idx]  (= Bt for P.V)
  __shared__ __align__(16) half_t Ps[8][16 * 32];  // per-wave P staging
  int h = blockIdx.x & 3, i = blockIdx.x >> 2;
  int tid = threadIdx.x, lane = tid & 31, w = tid >> 5;
  const half_t* prow = proj + (size_t)i * NRES * 512;

  // K tile: async 16B copies (4 chunks per row)
  for (int t = tid; t < NRES * 4; t += 256) {
    int kk = t >> 2, ch = (t & 3) * 8;
    cp_async16(&Ks[kk * 32 + ch], prow + (size_t)kk * 512 + 128 + h * 32 + ch);
  }
  // V^T tile: element transpose (plain loads)
  for (int t = tid; t < NRES * 32; t += 256) {
    int kk = t >> 5, c = t & 31;
    Vt[c * NRES + kk] = prow[(size_t)kk * 512 + 256 + h * 32 + c];
  }
  async_wait0();
  __syncthreads();

  int half_ = lane >> 4, ln = lane & 15;
  int j0 = w * 32;
  const float sc = 0.17677669529663687f;  // 1/sqrt(32)
  const v8f vz = {0.f, 0.f, 0.f, 0.f, 0.f, 0.f, 0.f, 0.f};

  for (int mc = 0; mc < 2; ++mc) {
    int jrow0 = j0 + mc * 16;
    v16h qa = load_a_frag(prow + (size_t)jrow0 * 512 + h * 32, 512);
    v8f oacc0 = vz, oacc1 = vz;
    float rmax[8], rsum[8];
#pragma unroll
    for (int r = 0; r < 8; ++r) { rmax[r] = -1e30f; rsum[r] = 0.f; }

    for (int kt = 0; kt < 8; ++kt) {
      int kbase = kt * 32;
      v16h bk0 = load_b_frag_nk(Ks + (size_t)kbase * 32, 32);
      v8f s0 = wmma16(qa, bk0, vz);
      v16h bk1 = load_b_frag_nk(Ks + (size_t)(kbase + 16) * 32, 32);
      v8f s1 = wmma16(qa, bk1, vz);

#pragma unroll
      for (int r = 0; r < 8; ++r) {
        int j = jrow0 + r + 8 * half_;
        float p0 = s0[r] * sc + biasw[(size_t)h * M_TOT + (size_t)j * NRES + kbase + ln];
        float p1 = s1[r] * sc + biasw[(size_t)h * M_TOT + (size_t)j * NRES + kbase + 16 + ln];
        float mx = fmaxf(p0, p1);
#pragma unroll
        for (int d = 1; d < 16; d <<= 1) mx = fmaxf(mx, __shfl_xor(mx, d, 32));
        float mnew = fmaxf(rmax[r], mx);
        float f = __expf(rmax[r] - mnew);
        float e0 = __expf(p0 - mnew), e1 = __expf(p1 - mnew);
        float ps = e0 + e1;
#pragma unroll
        for (int d = 1; d < 16; d <<= 1) ps += __shfl_xor(ps, d, 32);
        rsum[r] = rsum[r] * f + ps;
        rmax[r] = mnew;
        oacc0[r] *= f;
        oacc1[r] *= f;
        Ps[w][(r + 8 * half_) * 32 + ln]      = (half_t)e0;
        Ps[w][(r + 8 * half_) * 32 + 16 + ln] = (half_t)e1;
      }
      // same-wave LDS RAW; make the dependency explicit
      asm volatile("s_wait_dscnt 0" ::: "memory");
      v16h pa = load_a_frag(&Ps[w][0], 32);
      v16h bv0 = load_b_frag_nk(Vt + 0 * NRES + kbase, NRES);
      oacc0 = wmma16(pa, bv0, oacc0);
      v16h bv1 = load_b_frag_nk(Vt + 16 * NRES + kbase, NRES);
      oacc1 = wmma16(pa, bv1, oacc1);
    }

#pragma unroll
    for (int r = 0; r < 8; ++r) {
      int j = jrow0 + r + 8 * half_;
      float inv = 1.f / rsum[r];
      float g0 = (float)prow[(size_t)j * 512 + 384 + h * 32 + ln];
      float g1 = (float)prow[(size_t)j * 512 + 384 + h * 32 + 16 + ln];
      size_t base = ((size_t)i * NRES + j) * 128 + h * 32;
      go[base + ln]      = (half_t)(g0 * oacc0[r] * inv);
      go[base + 16 + ln] = (half_t)(g1 * oacc1[r] * inv);
    }
  }
}

// --------------------------- triangle multiplication ------------------------
// a = sigmoid(ag)*ap, b = sigmoid(bg)*bp; pack to channel-major slabs
// (transpose=1 writes slab transposed => implements the 'incoming' einsum)
__global__ __launch_bounds__(256)
void mult_ab_pack(const float* __restrict__ tmp, half_t* __restrict__ At,
                  half_t* __restrict__ Bt, int transpose) {
  int t = blockIdx.x * 256 + threadIdx.x;  // M_TOT*32
  int m = t >> 5, c = t & 31;
  const float* r = tmp + (size_t)m * 128;
  float a = sigmoidf_(r[c])      * r[32 + c];
  float b = sigmoidf_(r[64 + c]) * r[96 + c];
  int mi = m >> 8, mj = m & 255;
  int dst = transpose ? (mj * 256 + mi) : m;
  At[(size_t)c * M_TOT + dst] = (half_t)a;
  Bt[(size_t)c * M_TOT + dst] = (half_t)b;
}

// z[i,j,c] = sum_k At[c][i][k] * Bt[c][j][k]  (32 batched 256^3 GEMMs)
__global__ __launch_bounds__(256)
void tri_mult_gemm(const half_t* __restrict__ At, const half_t* __restrict__ Bt,
                   float* __restrict__ z32) {
  __shared__ __align__(16) half_t As[64 * 32];
  __shared__ __align__(16) half_t Bs[64 * 32];
  int c = blockIdx.x & 31;
  int ij = blockIdx.x >> 5;
  int i0 = (ij >> 2) * 64, j0 = (ij & 3) * 64;
  int tid = threadIdx.x, lane = tid & 31, w = tid >> 5;
  int wm = w & 3, wn = w >> 2;
  int m0 = wm * 16;
  const half_t* Ac = At + (size_t)c * M_TOT;
  const half_t* Bc = Bt + (size_t)c * M_TOT;
  v8f acc0 = {0.f, 0.f, 0.f, 0.f, 0.f, 0.f, 0.f, 0.f};
  v8f acc1 = acc0;
  int row = tid >> 2, off = (tid & 3) * 8;
  for (int k = 0; k < 256; k += 32) {
    cp_async16(&As[row * 32 + off], Ac + (size_t)(i0 + row) * 256 + k + off);
    cp_async16(&Bs[row * 32 + off], Bc + (size_t)(j0 + row) * 256 + k + off);
    async_wait0();
    __syncthreads();
    v16h a = load_a_frag(As + m0 * 32, 32);
    v16h b0 = load_b_frag_nk(Bs + (wn * 32 + 0) * 32, 32);
    acc0 = wmma16(a, b0, acc0);
    v16h b1 = load_b_frag_nk(Bs + (wn * 32 + 16) * 32, 32);
    acc1 = wmma16(a, b1, acc1);
    __syncthreads();
  }
  int half_ = lane >> 4, ln = lane & 15;
#pragma unroll
  for (int r = 0; r < 8; ++r) {
    int gi = i0 + m0 + r + 8 * half_;
    int gj = j0 + wn * 32 + ln;
    z32[((size_t)gi * 256 + gj) * 32 + c]      = acc0[r];
    z32[((size_t)gi * 256 + gj + 16) * 32 + c] = acc1[r];
  }
}

// ============================== host orchestration ==========================

// tree_leaves order (keys sorted at every level):
enum {
  AE_BG = 0, AE_BO, AE_LNB, AE_LNG, AE_WB, AE_WG, AE_WK, AE_WO, AE_WQ, AE_WV,
  AS_BG = 10, AS_BO, AS_LNB, AS_LNG, AS_WB, AS_WG, AS_WK, AS_WO, AS_WQ, AS_WV,
  FLN_B = 20, FLN_G = 21,
  MI_BAG = 22, MI_BAP, MI_BBG, MI_BBP, MI_BG, MI_BO, MI_LN2B, MI_LN2G, MI_LNB, MI_LNG,
  MI_WAG = 32, MI_WAP, MI_WBG, MI_WBP, MI_WG, MI_WO,
  MO_BAG = 38, MO_BAP, MO_BBG, MO_BBP, MO_BG, MO_BO, MO_LN2B, MO_LN2G, MO_LNB, MO_LNG,
  MO_WAG = 48, MO_WAP, MO_WBG, MO_WBP, MO_WG, MO_WO,
  T_B1 = 54, T_B2, T_LNB, T_LNG, T_W1, T_W2,
  IN_X = 60
};

extern "C" void kernel_launch(void* const* d_in, const int* in_sizes, int n_in,
                              void* d_out, int out_size, void* d_ws, size_t ws_size,
                              hipStream_t stream) {
  (void)in_sizes; (void)n_in; (void)out_size; (void)ws_size;
  auto F = [&](int i) { return (const float*)d_in[i]; };

  char* ws = (char*)d_ws;
  size_t off = 0;
  auto alloc = [&](size_t bytes) {
    void* p = ws + off;
    off = (off + bytes + 255) & ~(size_t)255;
    return p;
  };

  // persistent
  half_t* xl16    = (half_t*)alloc((size_t)M_TOT * 64 * 2);
  half_t* Wqkvg_s = (half_t*)alloc(512 * 64 * 2);
  half_t* Wqkvg_e = (half_t*)alloc(512 * 64 * 2);
  half_t* Wo_s    = (half_t*)alloc(64 * 128 * 2);
  half_t* Wo_e    = (half_t*)alloc(64 * 128 * 2);
  half_t* Wab_o   = (half_t*)alloc(128 * 64 * 2);
  half_t* Wab_i   = (half_t*)alloc(128 * 64 * 2);
  float*  B128_o  = (float*)alloc(128 * 4);
  float*  B128_i  = (float*)alloc(128 * 4);
  half_t* Wg_o    = (half_t*)alloc(64 * 64 * 2);
  half_t* Wg_i    = (half_t*)alloc(64 * 64 * 2);
  half_t* Wom_o   = (half_t*)alloc(64 * 32 * 2);
  half_t* Wom_i   = (half_t*)alloc(64 * 32 * 2);
  half_t* W1t     = (half_t*)alloc(128 * 64 * 2);
  half_t* W2t     = (half_t*)alloc(64 * 128 * 2);

  size_t arena = (off + 4095) & ~(size_t)4095;
  // attention phase scratch
  half_t* proj16 = (half_t*)(ws + arena);                         // 67,108,864 B
  float*  biasw  = (float*)(ws + arena + 67108864ull);            //  1,048,576 B
  half_t* go16   = (half_t*)(ws + arena + 68157440ull);           // 16,777,216 B
  // tri-mult phase scratch (aliases attention scratch)
  float*  tmp128 = (float*)(ws + arena);                          // 33,554,432 B
  half_t* Amt    = (half_t*)(ws + arena + 33554432ull);           //  4,194,304 B
  half_t* Bmt    = (half_t*)(ws + arena + 37748736ull);           //  4,194,304 B
  float*  gate32 = (float*)(ws + arena + 41943040ull);            // 16,777,216 B
  float*  z32    = (float*)(ws + arena + 58720256ull);            //  8,388,608 B
  half_t* zl16   = (half_t*)(ws + arena + 67108864ull);           //  4,194,304 B
  // transition scratch
  half_t* h16    = (half_t*)(ws + arena);                         // 16,777,216 B

  float* xw = (float*)d_out;  // f32 residual buffer, finalized in-place

  // ---- pack weights (f32 -> f16 Bt layouts) ----
  pack_qkvg<<<128, 256, 0, stream>>>(F(AS_WQ), F(AS_WK), F(AS_WV), F(AS_WG), Wqkvg_s);
  pack_qkvg<<<128, 256, 0, stream>>>(F(AE_WQ), F(AE_WK), F(AE_WV), F(AE_WG), Wqkvg_e);
  pack_wo_attn<<<32, 256, 0, stream>>>(F(AS_WO), Wo_s);
  pack_wo_attn<<<32, 256, 0, stream>>>(F(AE_WO), Wo_e);
  pack_ab<<<32, 256, 0, stream>>>(F(MO_WAG), F(MO_WAP), F(MO_WBG), F(MO_WBP),
                                  F(MO_BAG), F(MO_BAP), F(MO_BBG), F(MO_BBP), Wab_o, B128_o);
  pack_ab<<<32, 256, 0, stream>>>(F(MI_WAG), F(MI_WAP), F(MI_WBG), F(MI_WBP),
                                  F(MI_BAG), F(MI_BAP), F(MI_BBG), F(MI_BBP), Wab_i, B128_i);
  transpose_pack<<<16, 256, 0, stream>>>(F(MO_WG), Wg_o, 64, 64);
  transpose_pack<<<16, 256, 0, stream>>>(F(MI_WG), Wg_i, 64, 64);
  transpose_pack<<<8, 256, 0, stream>>>(F(MO_WO), Wom_o, 32, 64);
  transpose_pack<<<8, 256, 0, stream>>>(F(MI_WO), Wom_i, 32, 64);
  transpose_pack<<<32, 256, 0, stream>>>(F(T_W1), W1t, 64, 128);
  transpose_pack<<<32, 256, 0, stream>>>(F(T_W2), W2t, 128, 64);

  // ---- residual init: xw = x ----
  (void)hipMemcpyAsync(xw, d_in[IN_X], (size_t)M_TOT * 64 * 4,
                       hipMemcpyDeviceToDevice, stream);

  // ---- two triangle attentions: start (T=0) then end (T=1, transposed domain)
  for (int T = 0; T < 2; ++T) {
    int base = T ? AE_BG : AS_BG;
    half_t* Wqkvg = T ? Wqkvg_e : Wqkvg_s;
    half_t* Wo    = T ? Wo_e : Wo_s;
    ln64_f16<<<8192, 256, 0, stream>>>(xw, F(base + 3), F(base + 2), xl16, T);
    // qkv projection (Nout=384), then sigmoid-gate projection (Nout=128)
    gemm_nk<64, 384, F_OUT16><<<512, 256, 0, stream>>>(
        xl16, Wqkvg, nullptr, proj16, nullptr, nullptr, 512, 0);
    gemm_nk<64, 128, F_OUT16 | F_BIAS | F_SIG><<<512, 256, 0, stream>>>(
        xl16, Wqkvg + 384 * 64, F(base + 0), proj16, nullptr, nullptr, 512, 384);
    bias_proj<<<256, 256, 0, stream>>>(xl16, F(base + 4), biasw);
    tri_attn_fa<<<NH * NRES, 256, 0, stream>>>(proj16, biasw, go16);
    if (T == 0)
      gemm_nk<128, 64, F_BIAS | F_RES><<<512, 256, 0, stream>>>(
          go16, Wo, F(base + 1), nullptr, xw, nullptr, 64, 0);
    else
      gemm_nk<128, 64, F_BIAS | F_RES | F_REST><<<512, 256, 0, stream>>>(
          go16, Wo, F(base + 1), nullptr, xw, nullptr, 64, 0);
  }

  // ---- two triangle multiplications: outgoing (T=0), incoming (T=1)
  for (int T = 0; T < 2; ++T) {
    int base = T ? MI_BAG : MO_BAG;
    half_t* Wab = T ? Wab_i : Wab_o;
    float*  B128 = T ? B128_i : B128_o;
    half_t* Wg = T ? Wg_i : Wg_o;
    half_t* Wom = T ? Wom_i : Wom_o;
    ln64_f16<<<8192, 256, 0, stream>>>(xw, F(base + 9), F(base + 8), xl16, 0);
    gemm_nk<64, 128, F_BIAS | F_OUT32><<<512, 256, 0, stream>>>(
        xl16, Wab, B128, nullptr, tmp128, nullptr, 128, 0);
    mult_ab_pack<<<8192, 256, 0, stream>>>(tmp128, Amt, Bmt, T);
    gemm_nk<64, 64, F_BIAS | F_SIG | F_OUT32><<<512, 256, 0, stream>>>(
        xl16, Wg, F(base + 4), nullptr, gate32, nullptr, 64, 0);
    tri_mult_gemm<<<512, 256, 0, stream>>>(Amt, Bmt, z32);
    ln32_f16<<<8192, 256, 0, stream>>>(z32, F(base + 7), F(base + 6), zl16);
    gemm_nk<32, 64, F_BIAS | F_GATE | F_RES><<<512, 256, 0, stream>>>(
        zl16, Wom, F(base + 5), nullptr, xw, gate32, 64, 0);
  }

  // ---- transition MLP ----
  ln64_f16<<<8192, 256, 0, stream>>>(xw, F(T_LNG), F(T_LNB), xl16, 0);
  gemm_nk<64, 128, F_BIAS | F_RELU | F_OUT16><<<512, 256, 0, stream>>>(
      xl16, W1t, F(T_B1), h16, nullptr, nullptr, 128, 0);
  gemm_nk<128, 64, F_BIAS | F_RES><<<512, 256, 0, stream>>>(
      h16, W2t, F(T_B2), nullptr, xw, nullptr, 64, 0);

  // ---- final LayerNorm (in-place on d_out) ----
  final_ln64<<<8192, 256, 0, stream>>>(xw, F(FLN_G), F(FLN_B));
}
